// GAT_14766097564097
// MI455X (gfx1250) — matrix operand order repaired and checked by reading, not compiled, not run on hardware
//
#include <hip/hip_runtime.h>

#define B_ 8
#define N_ 1024
#define F_ 512
#define H_ 4
#define D_ 128
#define ALPHA_ 0.2f

typedef __attribute__((ext_vector_type(16))) __bf16 v16bf;
typedef __attribute__((ext_vector_type(8)))  __bf16 v8bf;
typedef __attribute__((ext_vector_type(4)))  __bf16 v4bf;
typedef __attribute__((ext_vector_type(8)))  float  v8f;
typedef __attribute__((ext_vector_type(4)))  float  v4f;

// ---- Stage 1a: u[h*2+s][f] = sum_d W_att[h][f][d] * a_att[h][s][d] ----
__global__ void gat_compute_u(const float* __restrict__ W_att,
                              const float* __restrict__ a_att,
                              float* __restrict__ u_all) {
  int t = blockIdx.x * blockDim.x + threadIdx.x;
  if (t >= H_ * 2 * F_) return;
  int f  = t % F_;
  int hs = t / F_;
  int h = hs >> 1, s = hs & 1;
  const float* w = W_att + ((size_t)h * F_ + f) * D_;
  const float* a = a_att + ((size_t)(h * 2 + s)) * D_;
  float acc = 0.f;
  #pragma unroll 8
  for (int d = 0; d < D_; ++d) acc += w[d] * a[d];
  u_all[hs * F_ + f] = acc;
}

// ---- Stage 1b: KT[h][d][f] = bf16(kernel[h][f][d]) ----
__global__ void gat_transpose_kernel(const float* __restrict__ kern,
                                     __bf16* __restrict__ KT) {
  int t = blockIdx.x * blockDim.x + threadIdx.x;  // H*D*F threads
  int f = t % F_;
  int d = (t / F_) % D_;
  int h = t / (F_ * D_);
  KT[t] = (__bf16)kern[((size_t)h * F_ + f) * D_ + d];
}

// ---- Stage 1c: Xb = bf16(X), row-major [B*N][F] ----
__global__ void gat_convert_x(const float* __restrict__ X, __bf16* __restrict__ Xb) {
  int t = blockIdx.x * blockDim.x + threadIdx.x;  // (B*N*F)/4 threads
  v4f x = ((const v4f*)X)[t];
  v4bf y;
  y[0] = (__bf16)x[0]; y[1] = (__bf16)x[1]; y[2] = (__bf16)x[2]; y[3] = (__bf16)x[3];
  ((v4bf*)Xb)[t] = y;
}

// ---- Stage 2: att_self/att_neigh [h][b*n] = X row dot u ----
__global__ void gat_att_scores(const float* __restrict__ X,
                               const float* __restrict__ u_all,
                               float* __restrict__ att_self,
                               float* __restrict__ att_neigh) {
  __shared__ float su[2 * H_ * F_];  // 16 KB
  for (int i = threadIdx.x; i < 2 * H_ * F_; i += 256) su[i] = u_all[i];
  __syncthreads();
  int wave = threadIdx.x >> 5, lane = threadIdx.x & 31;
  int row = blockIdx.x * 8 + wave;        // [0, B*N)
  const float* x = X + (size_t)row * F_;
  float acc[8] = {0.f, 0.f, 0.f, 0.f, 0.f, 0.f, 0.f, 0.f};
  for (int t = 0; t < F_ / 32; ++t) {
    int f = lane + t * 32;
    float xv = x[f];
    #pragma unroll
    for (int hs = 0; hs < 8; ++hs) acc[hs] += xv * su[hs * F_ + f];
  }
  #pragma unroll
  for (int hs = 0; hs < 8; ++hs) {
    float v = acc[hs];
    for (int off = 16; off >= 1; off >>= 1) v += __shfl_xor(v, off, 32);
    acc[hs] = v;
  }
  if (lane == 0) {
    #pragma unroll
    for (int h = 0; h < H_; ++h) {
      att_self [h * (B_ * N_) + row] = acc[h * 2 + 0];
      att_neigh[h * (B_ * N_) + row] = acc[h * 2 + 1];
    }
  }
}

// ---- Stage 3: Mx[h*B+b] = max_j att_neigh[h][b][j] ----
__global__ void gat_neigh_max(const float* __restrict__ att_neigh,
                              float* __restrict__ Mx) {
  __shared__ float sm[8];
  const float* p = att_neigh + (size_t)blockIdx.x * N_;
  float m = -3.0e38f;
  for (int i = threadIdx.x; i < N_; i += 256) m = fmaxf(m, p[i]);
  for (int off = 16; off >= 1; off >>= 1) m = fmaxf(m, __shfl_xor(m, off, 32));
  int wave = threadIdx.x >> 5, lane = threadIdx.x & 31;
  if (lane == 0) sm[wave] = m;
  __syncthreads();
  if (threadIdx.x == 0) {
    float r = sm[0];
    #pragma unroll
    for (int w = 1; w < 8; ++w) r = fmaxf(r, sm[w]);
    Mx[blockIdx.x] = r;
  }
}

// ---- Stage 4: XkT[h][b][d][node] = bf16( X @ kernel[h] )^T via WMMA ----
__global__ void __launch_bounds__(256)
gat_project(const __bf16* __restrict__ Xb, const __bf16* __restrict__ KT,
            __bf16* __restrict__ XkT) {
  int blk   = blockIdx.x;          // (h*B + b)*64 + mtile
  int mtile = blk & 63;
  int hb    = blk >> 6;
  int b = hb & 7, h = hb >> 3;
  int wave = threadIdx.x >> 5, lane = threadIdx.x & 31;
  int dt = wave;                   // one 16-wide d-tile per wave (D=128 -> 8 waves)
  int m0 = mtile * 16;
  int lhalf = lane >> 4, lmod = lane & 15;

  const __bf16* xrow = Xb + ((size_t)(b * N_) + m0 + lmod) * F_;
  const __bf16* krow = KT + ((size_t)(h * D_) + dt * 16 + lmod) * F_;
  v8f acc = {};
  for (int ft = 0; ft < F_ / 32; ++ft) {
    int f0 = ft * 32;
    v16bf afrag = *(const v16bf*)(xrow + f0 + 16 * lhalf);   // K contiguous
    v16bf bfrag = *(const v16bf*)(krow + f0 + 16 * lhalf);   // K contiguous
    acc = __builtin_amdgcn_wmma_f32_16x16x32_bf16(false, afrag, false, bfrag,
                                                  (short)0, acc, false, false);
  }
  // D-frag: lane holds (m = v + 8*lhalf, n = lmod); store transposed [d][node]
  __bf16* out = XkT + (((size_t)hb * D_) + dt * 16 + lmod) * N_ + m0 + 8 * lhalf;
  v8bf o;
  #pragma unroll
  for (int v = 0; v < 8; ++v) o[v] = (__bf16)acc[v];
  *(v8bf*)out = o;
}

// ---- Stage 5: fused masked-softmax attention + aggregate + bias + ELU ----
__global__ void __launch_bounds__(128)
gat_attend(const float* __restrict__ A, const float* __restrict__ att_self,
           const float* __restrict__ att_neigh, const float* __restrict__ Mx,
           const __bf16* __restrict__ XkT, const float* __restrict__ bias,
           float* __restrict__ out) {
  int blk = blockIdx.x;            // b*64 + itile
  int itile = blk & 63;
  int b = blk >> 6;
  int h = threadIdx.x >> 5;        // one head per wave
  int lane = threadIdx.x & 31;
  int lhalf = lane >> 4, lmod = lane & 15;
  int i0 = itile * 16;
  int hb = h * B_ + b;

  float self = att_self[(size_t)hb * N_ + i0 + lmod];  // i = i0 + lmod for this lane
  float mx = Mx[hb];
  float sm = self + mx;
  float m_i = sm >= 0.f ? sm : ALPHA_ * sm;            // safe per-row upper bound

  const float*  nrow = att_neigh + (size_t)hb * N_;
  const float*  arow = A + ((size_t)(b * N_) + i0 + lmod) * N_;
  const __bf16* xk   = XkT + (size_t)hb * D_ * N_;

  v8f acc[8];
  #pragma unroll
  for (int dt = 0; dt < 8; ++dt) acc[dt] = (v8f){};
  float sump = 0.f;

  for (int jt = 0; jt < N_ / 32; ++jt) {
    int j0 = jt * 32;
    int c1 = j0 + 16 * lhalf;      // 16 contiguous K elements
    v4f n0 = *(const v4f*)(nrow + c1);
    v4f n1 = *(const v4f*)(nrow + c1 + 4);
    v4f n2 = *(const v4f*)(nrow + c1 + 8);
    v4f n3 = *(const v4f*)(nrow + c1 + 12);
    v4f a0 = *(const v4f*)(arow + c1);
    v4f a1 = *(const v4f*)(arow + c1 + 4);
    v4f a2 = *(const v4f*)(arow + c1 + 8);
    v4f a3 = *(const v4f*)(arow + c1 + 12);

    v16bf p;
    #pragma unroll
    for (int e = 0; e < 16; ++e) {
      float nv = (e < 4) ? n0[e] : (e < 8) ? n1[e - 4] : (e < 12) ? n2[e - 8] : n3[e - 12];
      float av = (e < 4) ? a0[e] : (e < 8) ? a1[e - 4] : (e < 12) ? a2[e - 8] : a3[e - 12];
      float s = self + nv;
      float lr = s >= 0.f ? s : ALPHA_ * s;
      float pe = av * __expf(lr - m_i);   // masked entries -> exact 0
      __bf16 pb = (__bf16)pe;
      p[e] = pb;
      sump += (float)pb;                  // normalize with the rounded values
    }
    const __bf16* xb = xk + j0 + 16 * lhalf;
    #pragma unroll
    for (int dt = 0; dt < 8; ++dt) {
      v16bf bfrag = *(const v16bf*)(xb + ((size_t)(dt * 16 + lmod)) * N_);
      acc[dt] = __builtin_amdgcn_wmma_f32_16x16x32_bf16(false, p, false, bfrag,
                                                        (short)0, acc[dt], false, false);
    }
  }

  // row sums: lanes l and l^16 cover complementary j-halves of row i=l%16
  sump += __shfl_xor(sump, 16, 32);
  float rs[8];
  #pragma unroll
  for (int v = 0; v < 8; ++v) rs[v] = __shfl(sump, v + 8 * lhalf, 32);

  #pragma unroll
  for (int dt = 0; dt < 8; ++dt) {
    int d = dt * 16 + lmod;
    float bv = bias[h * D_ + d];
    #pragma unroll
    for (int v = 0; v < 8; ++v) {
      float val = acc[dt][v] / rs[v] + bv;
      val = val > 0.f ? val : expm1f(val);                 // ELU (alpha=1)
      out[((size_t)(b * N_) + i0 + v + 8 * lhalf) * (H_ * D_) + h * D_ + d] = val;
    }
  }
}

extern "C" void kernel_launch(void* const* d_in, const int* in_sizes, int n_in,
                              void* d_out, int out_size, void* d_ws, size_t ws_size,
                              hipStream_t stream) {
  const float* X     = (const float*)d_in[0];
  const float* A     = (const float*)d_in[1];
  const float* W_att = (const float*)d_in[2];
  const float* a_att = (const float*)d_in[3];
  const float* kern  = (const float*)d_in[4];
  const float* bias  = (const float*)d_in[5];
  float* out = (float*)d_out;

  char* ws = (char*)d_ws;
  size_t off = 0;
  auto alloc = [&](size_t bytes) -> void* {
    off = (off + 255) & ~(size_t)255;
    void* p = ws + off;
    off += bytes;
    return p;
  };
  float*  u_all     = (float*) alloc((size_t)2 * H_ * F_ * 4);        // 16 KB
  __bf16* Xb        = (__bf16*)alloc((size_t)B_ * N_ * F_ * 2);       // 8 MB
  __bf16* KT        = (__bf16*)alloc((size_t)H_ * D_ * F_ * 2);       // 512 KB
  float*  att_self  = (float*) alloc((size_t)H_ * B_ * N_ * 4);       // 128 KB
  float*  att_neigh = (float*) alloc((size_t)H_ * B_ * N_ * 4);       // 128 KB
  float*  Mx        = (float*) alloc((size_t)H_ * B_ * 4);
  __bf16* XkT       = (__bf16*)alloc((size_t)H_ * B_ * D_ * N_ * 2);  // 8 MB
  (void)ws_size; (void)in_sizes; (void)n_in; (void)out_size;

  gat_compute_u       <<<(H_ * 2 * F_ + 255) / 256, 256, 0, stream>>>(W_att, a_att, u_all);
  gat_transpose_kernel<<<(H_ * D_ * F_) / 256,      256, 0, stream>>>(kern, KT);
  gat_convert_x       <<<(B_ * N_ * F_ / 4) / 256,  256, 0, stream>>>(X, Xb);
  gat_att_scores      <<<(B_ * N_) / 8,             256, 0, stream>>>(X, u_all, att_self, att_neigh);
  gat_neigh_max       <<<H_ * B_,                   256, 0, stream>>>(att_neigh, Mx);
  gat_project         <<<H_ * B_ * 64,              256, 0, stream>>>(Xb, KT, XkT);
  gat_attend          <<<B_ * 64,                   128, 0, stream>>>(A, att_self, att_neigh, Mx,
                                                                      XkT, bias, out);
}
